// BinaryNetFCBlock_79912161509543
// MI455X (gfx1250) — 8-bit matrix operand order repaired and checked by reading, compiled, not run on hardware
//
#include <hip/hip_runtime.h>
#include <hip/hip_bf16.h>
#include <stdint.h>

typedef __attribute__((ext_vector_type(16))) int   v16i;
typedef __attribute__((ext_vector_type(8)))  float v8f;

#define B_ROWS 8192
#define D_DIM  4096
#define N_DIM  4096
#define KT_CNT (D_DIM / 128)
#define A_RT   (B_ROWS / 16)
#define B_NT   (N_DIM / 16)
#define TILE_BYTES 2048

#define FP8_P1 0x38u
#define FP8_M1 0xB8u

typedef int v4i_ __attribute__((vector_size(16)));
typedef __attribute__((address_space(1))) v4i_ g_v4;
typedef __attribute__((address_space(3))) v4i_ l_v4;

__device__ __forceinline__ g_v4* as_g(const void* p) {
    return (g_v4*)(unsigned long long)(uintptr_t)p;
}
__device__ __forceinline__ l_v4* as_l(void* p) {
    return (l_v4*)(unsigned)(uintptr_t)p;
}

__device__ __forceinline__ void copy64_to_lds(const unsigned char* g, unsigned char* l) {
#if __has_builtin(__builtin_amdgcn_global_load_async_to_lds_b128)
    __builtin_amdgcn_global_load_async_to_lds_b128(as_g(g), as_l(l), 0, 0);
    __builtin_amdgcn_global_load_async_to_lds_b128(as_g(g), as_l(l), 16, 0);
    __builtin_amdgcn_global_load_async_to_lds_b128(as_g(g), as_l(l), 32, 0);
    __builtin_amdgcn_global_load_async_to_lds_b128(as_g(g), as_l(l), 48, 0);
#else
    const uint4* s = (const uint4*)g;
    uint4*       d = (uint4*)l;
#pragma unroll
    for (int i = 0; i < 4; ++i) d[i] = s[i];
#endif
}

__device__ __forceinline__ void wait_async_prev_stage() {
#if __has_builtin(__builtin_amdgcn_s_wait_asynccnt)
    __builtin_amdgcn_s_wait_asynccnt(8);
#elif __has_builtin(__builtin_amdgcn_global_load_async_to_lds_b128)
    asm volatile("s_wait_asynccnt 0x8" ::: "memory");
#endif
}
__device__ __forceinline__ void wait_async_all() {
#if __has_builtin(__builtin_amdgcn_s_wait_asynccnt)
    __builtin_amdgcn_s_wait_asynccnt(0);
#elif __has_builtin(__builtin_amdgcn_global_load_async_to_lds_b128)
    asm volatile("s_wait_asynccnt 0x0" ::: "memory");
#endif
}

__global__ __launch_bounds__(256)
void pack_a_kernel(const float* __restrict__ x, unsigned* __restrict__ aq) {
    unsigned idx  = blockIdx.x * blockDim.x + threadIdx.x;
    unsigned v    = idx & 15;
    unsigned lane = (idx >> 4) & 31;
    unsigned kt   = (idx >> 9) & (KT_CNT - 1);
    unsigned rt   = idx >> 14;
    if (rt >= A_RT) return;

    unsigned m  = lane & 15;
    unsigned hi = lane >> 4;
    unsigned v7 = v & 7;
    unsigned k0 = ((v7 & 1) << 2) | (hi << 3) | (((v7 >> 1) & 1) << 4) |
                  (((v7 >> 2) & 1) << 5) | ((v >> 3) << 6);
    unsigned row   = rt * 16 + m;
    unsigned kbase = kt * 128 + k0;

    const float* src = x + (size_t)row * D_DIM + kbase;
    unsigned out = 0;
#pragma unroll
    for (int b = 0; b < 4; ++b) {
        unsigned f8 = (src[b] >= 0.0f) ? FP8_P1 : FP8_M1;
        out |= f8 << (8 * b);
    }
    size_t tile = (size_t)rt * KT_CNT + kt;
    aq[tile * (TILE_BYTES / 4) + lane * 16 + v] = out;
}

__global__ __launch_bounds__(256)
void pack_b_kernel(const float* __restrict__ w, unsigned* __restrict__ bq) {
    unsigned idx  = blockIdx.x * blockDim.x + threadIdx.x;
    unsigned v    = idx & 15;
    unsigned lane = (idx >> 4) & 31;
    unsigned kt   = (idx >> 9) & (KT_CNT - 1);
    unsigned nt   = idx >> 14;
    if (nt >= B_NT) return;

    unsigned n  = nt * 16 + (lane & 15);
    unsigned hi = lane >> 4;
    unsigned k0 = ((v & 1) << 2) | (hi << 3) | (((v >> 1) & 1) << 4) | ((v >> 2) << 5);
    unsigned kbase = kt * 128 + k0;

    const float* src = w + (size_t)kbase * N_DIM + n;
    unsigned out = 0;
#pragma unroll
    for (int b = 0; b < 4; ++b) {
        unsigned f8 = (src[(size_t)b * N_DIM] >= 0.0f) ? FP8_P1 : FP8_M1;
        out |= f8 << (8 * b);
    }
    size_t tile = (size_t)nt * KT_CNT + kt;
    bq[tile * (TILE_BYTES / 4) + lane * 16 + v] = out;
}

__global__ __launch_bounds__(256, 1)
void bgemm_bn_kernel(const unsigned char* __restrict__ aq,
                     const unsigned char* __restrict__ bq,
                     const float* __restrict__ beta,
                     const float* __restrict__ mean,
                     const float* __restrict__ var,
                     float* __restrict__ out) {
    __shared__ unsigned char ldsA[2][8 * TILE_BYTES];
    __shared__ unsigned char ldsB[2][8 * TILE_BYTES];

    const int tid   = threadIdx.x;
    const int lane  = tid & 31;
    const int wave  = tid >> 5;
    const int chunk = wave;
    const int wm    = wave >> 1;
    const int wn    = wave & 1;

    const unsigned char* gA =
        aq + ((size_t)(blockIdx.y * 8 + chunk) * KT_CNT) * TILE_BYTES + lane * 64;
    const unsigned char* gB =
        bq + ((size_t)(blockIdx.x * 8 + chunk) * KT_CNT) * TILE_BYTES + lane * 64;
    const unsigned ldst = chunk * TILE_BYTES + lane * 64;

    auto issue_stage = [&](int kt, int buf) {
        copy64_to_lds(gA + (size_t)kt * TILE_BYTES, &ldsA[buf][ldst]);
        copy64_to_lds(gB + (size_t)kt * TILE_BYTES, &ldsB[buf][ldst]);
    };

    issue_stage(0, 0);
    issue_stage(1, 1);

    v8f acc[2][4] = {};
    const unsigned aoff = (wm * 2) * TILE_BYTES + lane * 64;
    const unsigned boff = (wn * 4) * TILE_BYTES + lane * 64;

    for (int kt = 0; kt < KT_CNT; ++kt) {
        if (kt + 1 < KT_CNT) wait_async_prev_stage();
        else                 wait_async_all();
        __syncthreads();

        const unsigned char* pa = &ldsA[kt & 1][0];
        const unsigned char* pb = &ldsB[kt & 1][0];

        v16i a[2];
#pragma unroll
        for (int i = 0; i < 2; ++i)
            a[i] = *(const v16i*)(pa + aoff + i * TILE_BYTES);
        v16i b[4];
#pragma unroll
        for (int j = 0; j < 4; ++j)
            b[j] = *(const v16i*)(pb + boff + j * TILE_BYTES);

#pragma unroll
        for (int i = 0; i < 2; ++i)
#pragma unroll
            for (int j = 0; j < 4; ++j)
                acc[i][j] = __builtin_amdgcn_wmma_f32_16x16x128_fp8_fp8(
                    a[i], b[j], (short)0, acc[i][j], false, false);

        __syncthreads();
        if (kt + 2 < KT_CNT) issue_stage(kt + 2, kt & 1);
    }

    const int hi    = lane >> 4;
    const int cin   = lane & 15;
    const int row00 = blockIdx.y * 128 + wm * 32 + hi * 8;
    const int col00 = blockIdx.x * 128 + wn * 64 + cin;

#pragma unroll
    for (int j = 0; j < 4; ++j) {
        const int col = col00 + j * 16;
        const float inv = rsqrtf(var[col] + 1e-3f);
        const float mu  = mean[col];
        const float bt  = beta[col];
#pragma unroll
        for (int i = 0; i < 2; ++i) {
            const int rb = row00 + i * 16;
#pragma unroll
            for (int r = 0; r < 8; ++r) {
                out[(size_t)(rb + r) * N_DIM + col] = (acc[i][j][r] - mu) * inv + bt;
            }
        }
    }
}

extern "C" void kernel_launch(void* const* d_in, const int* in_sizes, int n_in,
                              void* d_out, int out_size, void* d_ws, size_t ws_size,
                              hipStream_t stream) {
    const float* x    = (const float*)d_in[0];
    const float* w    = (const float*)d_in[1];
    const float* beta = (const float*)d_in[2];
    const float* mean = (const float*)d_in[3];
    const float* var  = (const float*)d_in[4];
    float* out = (float*)d_out;

    unsigned char* aq = (unsigned char*)d_ws;
    unsigned char* bq = (unsigned char*)d_ws + (size_t)B_ROWS * D_DIM;

    {
        const unsigned total = A_RT * KT_CNT * 32 * 16;
        pack_a_kernel<<<total / 256, 256, 0, stream>>>(x, (unsigned*)aq);
    }
    {
        const unsigned total = B_NT * KT_CNT * 32 * 16;
        pack_b_kernel<<<total / 256, 256, 0, stream>>>(w, (unsigned*)bq);
    }
    {
        dim3 grid(N_DIM / 128, B_ROWS / 128);
        bgemm_bn_kernel<<<grid, 256, 0, stream>>>(aq, bq, beta, mean, var, out);
    }
}
